// LSHashNGP_43619687858999
// MI455X (gfx1250) — matrix operand order repaired and checked by reading, not compiled, not run on hardware
//
#include <hip/hip_runtime.h>

typedef __attribute__((ext_vector_type(16))) _Float16 v16h;
typedef __attribute__((ext_vector_type(8)))  float    v8f;
typedef __attribute__((ext_vector_type(4)))  int      v4i;

#define NPTS   2097152
#define NHEAD  16
#define TABM   524288
#define NTILES (NPTS / 16)   // 2^17

// ---- weight fragment loaders (setup only, amortized over ~32 tiles/wave) ----

// A-fragment of W^T consuming NATURAL feature order (layer 1 only):
// m = 16t + (lane&15); element j -> k = kbase + 16*s + j
__device__ __forceinline__ v16h load_a_frag(const float* __restrict__ W, int d2,
                                            int t, int kbase, int lane) {
  const int m = 16 * t + (lane & 15);
  const int s = lane >> 4;
  v16h r;
#pragma unroll
  for (int j = 0; j < 16; ++j) {
    const int k = kbase + 16 * s + j;
    r[j] = (_Float16)W[(size_t)k * d2 + m];
  }
  return r;
}

// A-fragment consuming the PERMUTED activation order produced by transition():
// element j -> k_feat = kbase + 16*(j>>3) + 8*s + (j&7)
__device__ __forceinline__ v16h load_a_frag_perm(const float* __restrict__ W, int d2,
                                                 int t, int kbase, int lane) {
  const int m = 16 * t + (lane & 15);
  const int s = lane >> 4;
  v16h r;
#pragma unroll
  for (int j = 0; j < 16; ++j) {
    const int k = kbase + 16 * (j >> 3) + 8 * s + (j & 7);
    r[j] = (_Float16)W[(size_t)k * d2 + m];
  }
  return r;
}

// Permuted-order consumer with zero-padded rows (W4: 64x3 -> 16 padded rows)
__device__ __forceinline__ v16h load_a_frag_pad_perm(const float* __restrict__ W, int d2,
                                                     int rows, int kbase, int lane) {
  const int m = lane & 15;
  const int s = lane >> 4;
  v16h r;
#pragma unroll
  for (int j = 0; j < 16; ++j) {
    const int k = kbase + 16 * (j >> 3) + 8 * s + (j & 7);
    r[j] = (m < rows) ? (_Float16)W[(size_t)k * d2 + m] : (_Float16)0.0f;
  }
  return r;
}

// Bias as C-matrix fragment: vgpr r holds row m = 16t + 8*(lane>>4) + r (all columns)
__device__ __forceinline__ v8f load_bias_frag(const float* __restrict__ b, int t,
                                              int lane, int dim) {
  const int base = 16 * t + 8 * (lane >> 4);
  v8f r;
#pragma unroll
  for (int j = 0; j < 8; ++j) r[j] = (base + j < dim) ? b[base + j] : 0.0f;
  return r;
}

// Pinned f32x2 -> packed f16 (stops clang from batching cvts + re-pairing with v_perm)
__device__ __forceinline__ unsigned cvt_pk(float x, float y) {
  unsigned r;
  asm("v_cvt_pk_f16_f32 %0, %1, %2" : "=v"(r) : "v"(x), "v"(y));
  return r;
}

// Fused f32x2 -> f16x2 convert + packed ReLU: exactly 2 VALU ops per dword.
__device__ __forceinline__ unsigned cvt_relu_pk(float x, float y) {
  unsigned r;
  asm("v_cvt_pk_f16_f32 %0, %1, %2\n\t"
      "v_pk_max_num_f16 %0, %0, 0"
      : "=v"(r) : "v"(x), "v"(y));
  return r;
}

// D-fragments (4 tiles) -> next-layer B-fragments (2 K-chunks), ReLU + f16 pack.
// Lane-local by construction: B[c] element j = relu(D[2c + (j>>3)][j&7]); the
// feature permutation this induces is compensated in load_a_frag_*perm above.
__device__ __forceinline__ void transition(const v8f D[4], v16h B[2]) {
#pragma unroll
  for (int c = 0; c < 2; ++c) {
    union { v16h v; unsigned u[8]; } o;
#pragma unroll
    for (int d = 0; d < 4; ++d) {
      o.u[d]     = cvt_relu_pk(D[2 * c][2 * d],     D[2 * c][2 * d + 1]);
      o.u[d + 4] = cvt_relu_pk(D[2 * c + 1][2 * d], D[2 * c + 1][2 * d + 1]);
    }
    B[c] = o.v;
  }
}

// ---- pipeline stages (all offsets fit u32; uniform-base + voffset form) ----
__device__ __forceinline__ void load_idx_pair(const char* __restrict__ idxb, int tile,
                                              unsigned lbase, v4i& a, v4i& b) {
  const v4i* ip = (const v4i*)(idxb + (((unsigned)tile << 10) + lbase));
  a = __builtin_nontemporal_load(ip);
  b = __builtin_nontemporal_load(ip + 1);
}

__device__ __forceinline__ void gather8(const char* __restrict__ tabb,
                                        const unsigned hbase[8],
                                        const v4i& a, const v4i& b, float2 f[8]) {
  unsigned id[8];
  id[0] = (unsigned)a.x; id[1] = (unsigned)a.y; id[2] = (unsigned)a.z; id[3] = (unsigned)a.w;
  id[4] = (unsigned)b.x; id[5] = (unsigned)b.y; id[6] = (unsigned)b.z; id[7] = (unsigned)b.w;
#pragma unroll
  for (int j = 0; j < 8; ++j)
    f[j] = *(const float2*)(tabb + (hbase[j] + (id[j] << 3)));
}

__device__ __forceinline__ v16h packX(const float2 f[8]) {
  union { v16h v; unsigned u[8]; } xb;
#pragma unroll
  for (int j = 0; j < 8; ++j) xb.u[j] = cvt_pk(f[j].x, f[j].y);
  return xb.v;
}

__global__ __launch_bounds__(256) void ngp_fused(
    const int* __restrict__ idx, const float* __restrict__ tables,
    const float* __restrict__ W1, const float* __restrict__ b1,
    const float* __restrict__ W2, const float* __restrict__ b2,
    const float* __restrict__ W3, const float* __restrict__ b3,
    const float* __restrict__ W4, const float* __restrict__ b4,
    float* __restrict__ out) {
  const int lane   = (int)(threadIdx.x & 31u);
  const int s      = lane >> 4;
  const int lane15 = lane & 15;
  const int wid    = (int)((blockIdx.x * blockDim.x + threadIdx.x) >> 5);
  const int nw     = (int)((gridDim.x * blockDim.x) >> 5);

  const char* idxb = (const char*)idx;
  const char* tabb = (const char*)tables;
  const unsigned lbase = (unsigned)(lane15 * (NHEAD * 4) + s * 32);  // idx row byte off
  const unsigned obase = (unsigned)(lane15 * 12);                    // out row byte off
  unsigned hbase[8];                                                 // head byte offsets
#pragma unroll
  for (int j = 0; j < 8; ++j) hbase[j] = (unsigned)(8 * s + j) << 22;  // *TABM*F*4

  // ---- resident weights/biases as WMMA fragments (per-wave registers) ----
  v16h A1[4], A2[4][2], A3[4][2], A4[2];
  v8f  C1[4], C2[4], C3[4], C4;
#pragma unroll
  for (int t = 0; t < 4; ++t) {
    A1[t]    = load_a_frag(W1, 64, t, 0, lane);          // natural (gathered) input
    A2[t][0] = load_a_frag_perm(W2, 64, t, 0, lane);     // permuted activations
    A2[t][1] = load_a_frag_perm(W2, 64, t, 32, lane);
    A3[t][0] = load_a_frag_perm(W3, 64, t, 0, lane);
    A3[t][1] = load_a_frag_perm(W3, 64, t, 32, lane);
    C1[t] = load_bias_frag(b1, t, lane, 64);
    C2[t] = load_bias_frag(b2, t, lane, 64);
    C3[t] = load_bias_frag(b3, t, lane, 64);
  }
  A4[0] = load_a_frag_pad_perm(W4, 3, 3, 0, lane);
  A4[1] = load_a_frag_pad_perm(W4, 3, 3, 32, lane);
  C4    = load_bias_frag(b4, 0, lane, 3);

  if (wid >= NTILES) return;

  // ---- software pipeline prologue ----
  // Xc   : features for current tile
  // ia/ib: idx values for tile+nw (feed next iteration's gathers)
  v16h Xc;
  v4i  ia, ib;
  {
    v4i a0, b0;
    load_idx_pair(idxb, wid, lbase, a0, b0);
    float2 f0[8];
    gather8(tabb, hbase, a0, b0, f0);
    Xc = packX(f0);
    load_idx_pair(idxb, (wid + nw) & (NTILES - 1), lbase, ia, ib);
  }

  for (int tile = wid; tile < NTILES; tile += nw) {
    // stage 1: stream idx for tile+2*nw (power-of-two wrap keeps it in-bounds;
    // wrapped prefetches are discarded). In flight through the whole MLP.
    v4i na, nb;
    load_idx_pair(idxb, (tile + 2 * nw) & (NTILES - 1), lbase, na, nb);

    // stage 2: issue gathers for tile+nw (consumed after the MLP)
    float2 fn[8];
    gather8(tabb, hbase, ia, ib, fn);

    // stage 3: fused MLP on current tile's features
    v8f D[4];
#pragma unroll
    for (int t = 0; t < 4; ++t)
      D[t] = __builtin_amdgcn_wmma_f32_16x16x32_f16(false, A1[t], false, Xc,
                                                    (short)0, C1[t], false, false);
    v16h B[2];
    transition(D, B);

#pragma unroll
    for (int t = 0; t < 4; ++t) {
      D[t] = __builtin_amdgcn_wmma_f32_16x16x32_f16(false, A2[t][0], false, B[0],
                                                    (short)0, C2[t], false, false);
      D[t] = __builtin_amdgcn_wmma_f32_16x16x32_f16(false, A2[t][1], false, B[1],
                                                    (short)0, D[t], false, false);
    }
    transition(D, B);

#pragma unroll
    for (int t = 0; t < 4; ++t) {
      D[t] = __builtin_amdgcn_wmma_f32_16x16x32_f16(false, A3[t][0], false, B[0],
                                                    (short)0, C3[t], false, false);
      D[t] = __builtin_amdgcn_wmma_f32_16x16x32_f16(false, A3[t][1], false, B[1],
                                                    (short)0, D[t], false, false);
    }
    transition(D, B);

    v8f O;
    O = __builtin_amdgcn_wmma_f32_16x16x32_f16(false, A4[0], false, B[0],
                                               (short)0, C4, false, false);
    O = __builtin_amdgcn_wmma_f32_16x16x32_f16(false, A4[1], false, B[1],
                                               (short)0, O, false, false);

    // lanes 0-15 hold m = 0..7 in vgprs 0..7; m<3 are the real outputs
    if (lane < 16) {
      float* op = (float*)((char*)out + ((unsigned)tile * 192u + obase));
      __builtin_nontemporal_store(O[0], op);
      __builtin_nontemporal_store(O[1], op + 1);
      __builtin_nontemporal_store(O[2], op + 2);
    }

    // stage 4: retire in-flight loads -> pipeline state for next iteration
    Xc = packX(fn);
    ia = na; ib = nb;
  }
}

extern "C" void kernel_launch(void* const* d_in, const int* in_sizes, int n_in,
                              void* d_out, int out_size, void* d_ws, size_t ws_size,
                              hipStream_t stream) {
  (void)in_sizes; (void)n_in; (void)out_size; (void)d_ws; (void)ws_size;
  const int*   idx    = (const int*)d_in[0];
  const float* tables = (const float*)d_in[1];
  const float* W1 = (const float*)d_in[2];
  const float* b1 = (const float*)d_in[3];
  const float* W2 = (const float*)d_in[4];
  const float* b2 = (const float*)d_in[5];
  const float* W3 = (const float*)d_in[6];
  const float* b3 = (const float*)d_in[7];
  const float* W4 = (const float*)d_in[8];
  const float* b4 = (const float*)d_in[9];

  dim3 grid(512), block(256);  // 4096 waves, 32 tiles of 16 points each
  hipLaunchKernelGGL(ngp_fused, grid, block, 0, stream,
                     idx, tables, W1, b1, W2, b2, W3, b3, W4, b4, (float*)d_out);
}
